// OneHopThenKConv_80032420593761
// MI455X (gfx1250) — matrix operand order repaired and checked by reading, not compiled, not run on hardware
//
#include <hip/hip_runtime.h>

typedef __attribute__((ext_vector_type(16))) _Float16 v16h;
typedef __attribute__((ext_vector_type(8)))  float    v8f;
typedef __attribute__((ext_vector_type(4)))  float    v4f;

union Frag {
  v16h v;
  _Float16 h[16];
  v4f f4[2];
};

struct H4 { _Float16 a, b, c, d; };

static __device__ __forceinline__ v8f wmma16(v16h a, v16h b, v8f c) {
  // D = A(16x32 f16) * B(32x16 f16) + C(16x16 f32)
  return __builtin_amdgcn_wmma_f32_16x16x32_f16(false, a, false, b, (short)0, c,
                                                false, false);
}

static __device__ __forceinline__ v8f splat8(float x) {
  v8f r = {x, x, x, x, x, x, x, x};
  return r;
}

static __device__ __forceinline__ void atomAdd(float* p, float v) {
  unsafeAtomicAdd(p, v);  // hardware global_atomic_add_f32
}

// ---------------------------------------------------------------- converters
__global__ void cvt_f16_k(const float* __restrict__ in, _Float16* __restrict__ out,
                          int n4) {
  int i = blockIdx.x * blockDim.x + threadIdx.x;
  if (i >= n4) return;
  v4f v = reinterpret_cast<const v4f*>(in)[i];
  H4 o = {(_Float16)v.x, (_Float16)v.y, (_Float16)v.z, (_Float16)v.w};
  reinterpret_cast<H4*>(out)[i] = o;
}

// W1 (272x128 f32) -> w1h (288x128 f16, rows 272..287 zero)
__global__ void pack_w1_k(const float* __restrict__ W1, _Float16* __restrict__ w1h) {
  int i = blockIdx.x * blockDim.x + threadIdx.x;
  if (i >= 288 * 128) return;
  int r = i >> 7;
  w1h[i] = (r < 272) ? (_Float16)W1[i] : (_Float16)0.f;
}

// ------------------------------------------------------------ degree / norm
__global__ void deg_k(const int* __restrict__ col, float* __restrict__ deg, int E) {
  int e = blockIdx.x * blockDim.x + threadIdx.x;
  if (e < E) atomAdd(deg + col[e], 1.0f);
}

__global__ void dis_k(float* __restrict__ degdis, int N) {  // in place deg -> D^-1/2
  int i = blockIdx.x * blockDim.x + threadIdx.x;
  if (i >= N) return;
  float d = degdis[i];
  degdis[i] = (d > 0.f) ? rsqrtf(fmaxf(d, 1.f)) : 0.f;
}

__global__ void norm_k(const int* __restrict__ row, const int* __restrict__ col,
                       const float* __restrict__ dis, float* __restrict__ nrm, int E) {
  int e = blockIdx.x * blockDim.x + threadIdx.x;
  if (e < E) nrm[e] = dis[row[e]] * dis[col[e]];
}

// ---------------------------------------------------------------------- SpMM
// out[col[e]] += norm[e] * in[row[e]] ; 32 threads per edge, 4 floats each
__global__ void spmm_k(const int* __restrict__ row, const int* __restrict__ col,
                       const float* __restrict__ nrm, const float* __restrict__ vin,
                       float* __restrict__ vout, int E) {
  int gid = blockIdx.x * blockDim.x + threadIdx.x;
  int e = gid >> 5;
  if (e >= E) return;
  int seg = gid & 31;
  float w = nrm[e];
  long rb = (long)row[e] * 128 + seg * 4;
  long cb = (long)col[e] * 128 + seg * 4;
  v4f v = *reinterpret_cast<const v4f*>(vin + rb);
  float* o = vout + cb;
  atomAdd(o + 0, w * v.x);
  atomAdd(o + 1, w * v.y);
  atomAdd(o + 2, w * v.z);
  atomAdd(o + 3, w * v.w);
}

// ------------------------------------------------------------- edge MLP (WMMA)
// per wave: 32 edges = two 16-row M-tiles sharing every B fragment (halves
// weight-cache traffic). GEMM1: [32 x 288] x [288 x 128] (chunks: 4 of x_i,
// 4 of x_j, 1 of edge_attr+pad). ReLU -> LDS(f16) -> GEMM2: [32x128]x[128x128].
// Scatter-add result rows into hacc[col[e]].
__global__ void __launch_bounds__(128)
edge_mlp_k(const _Float16* __restrict__ xh, const float* __restrict__ ea,
           const int* __restrict__ row, const int* __restrict__ col,
           const _Float16* __restrict__ w1h, const float* __restrict__ b1,
           const _Float16* __restrict__ w2h, const float* __restrict__ b2,
           float* __restrict__ hacc, int E) {
  __shared__ _Float16 lds[4][32 * 128];
  const int lane = threadIdx.x & 31;
  const int wave = threadIdx.x >> 5;
  const int tile = blockIdx.x * 4 + wave;
  const long eBase = (long)tile * 32;
  if (eBase >= E) return;
  const int m = lane & 15;
  const int hh = lane >> 4;
  long em0 = eBase + m;
  long em1 = eBase + 16 + m;
  if (em0 >= E) em0 = E - 1;  // clamped rows never scattered (guard below)
  if (em1 >= E) em1 = E - 1;
  const int ci0 = col[em0], ri0 = row[em0];
  const int ci1 = col[em1], ri1 = row[em1];

  v8f acc0[8], acc1[8];
#pragma unroll
  for (int t = 0; t < 8; ++t) {
    float bv = b1[t * 16 + m];
    acc0[t] = splat8(bv);
    acc1[t] = splat8(bv);
  }

  // ---- GEMM1: 9 K-chunks of 32
#pragma unroll
  for (int c = 0; c < 9; ++c) {
    Frag a0, a1;
    if (c < 8) {
      const _Float16* f0 = xh + (long)(c < 4 ? ci0 : ri0) * 128 + (c & 3) * 32;
      const _Float16* f1 = xh + (long)(c < 4 ? ci1 : ri1) * 128 + (c & 3) * 32;
      a0.f4[0] = *reinterpret_cast<const v4f*>(f0 + hh * 8);
      a0.f4[1] = *reinterpret_cast<const v4f*>(f0 + 16 + hh * 8);
      a1.f4[0] = *reinterpret_cast<const v4f*>(f1 + hh * 8);
      a1.f4[1] = *reinterpret_cast<const v4f*>(f1 + 16 + hh * 8);
    } else {
      // 16 edge-attr cols; upper 16 K of the chunk are zero pad
      const float* e0 = ea + em0 * 16 + hh * 8;
      const float* e1 = ea + em1 * 16 + hh * 8;
      v4f p0 = *reinterpret_cast<const v4f*>(e0);
      v4f p1 = *reinterpret_cast<const v4f*>(e0 + 4);
      v4f q0 = *reinterpret_cast<const v4f*>(e1);
      v4f q1 = *reinterpret_cast<const v4f*>(e1 + 4);
      a0.h[0] = (_Float16)p0.x; a0.h[1] = (_Float16)p0.y;
      a0.h[2] = (_Float16)p0.z; a0.h[3] = (_Float16)p0.w;
      a0.h[4] = (_Float16)p1.x; a0.h[5] = (_Float16)p1.y;
      a0.h[6] = (_Float16)p1.z; a0.h[7] = (_Float16)p1.w;
      a1.h[0] = (_Float16)q0.x; a1.h[1] = (_Float16)q0.y;
      a1.h[2] = (_Float16)q0.z; a1.h[3] = (_Float16)q0.w;
      a1.h[4] = (_Float16)q1.x; a1.h[5] = (_Float16)q1.y;
      a1.h[6] = (_Float16)q1.z; a1.h[7] = (_Float16)q1.w;
#pragma unroll
      for (int k = 8; k < 16; ++k) { a0.h[k] = (_Float16)0.f; a1.h[k] = (_Float16)0.f; }
    }
#pragma unroll
    for (int t = 0; t < 8; ++t) {
      const _Float16* wb = w1h + (long)(c * 32 + 8 * hh) * 128 + t * 16 + m;
      Frag b;
#pragma unroll
      for (int s = 0; s < 16; ++s) b.h[s] = wb[(long)((s & 7) + 16 * (s >> 3)) * 128];
      acc0[t] = wmma16(a0.v, b.v, acc0[t]);
      acc1[t] = wmma16(a1.v, b.v, acc1[t]);
    }
  }

  // ---- ReLU, stage hidden (32x128 f16) in per-wave LDS slice
#pragma unroll
  for (int t = 0; t < 8; ++t)
#pragma unroll
    for (int r = 0; r < 8; ++r) {
      lds[wave][(r + 8 * hh) * 128 + t * 16 + m] = (_Float16)fmaxf(acc0[t][r], 0.f);
      lds[wave][(16 + r + 8 * hh) * 128 + t * 16 + m] = (_Float16)fmaxf(acc1[t][r], 0.f);
    }

  // ---- GEMM2: K=128 (4 chunks), bias b2
#pragma unroll
  for (int t = 0; t < 8; ++t) {
    float bv = b2[t * 16 + m];
    acc0[t] = splat8(bv);
    acc1[t] = splat8(bv);
  }
#pragma unroll
  for (int c = 0; c < 4; ++c) {
    const _Float16* f0 = &lds[wave][m * 128 + c * 32];
    const _Float16* f1 = &lds[wave][(16 + m) * 128 + c * 32];
    Frag a0, a1;
    a0.f4[0] = *reinterpret_cast<const v4f*>(f0 + hh * 8);
    a0.f4[1] = *reinterpret_cast<const v4f*>(f0 + 16 + hh * 8);
    a1.f4[0] = *reinterpret_cast<const v4f*>(f1 + hh * 8);
    a1.f4[1] = *reinterpret_cast<const v4f*>(f1 + 16 + hh * 8);
#pragma unroll
    for (int t = 0; t < 8; ++t) {
      const _Float16* wb = w2h + (long)(c * 32 + 8 * hh) * 128 + t * 16 + m;
      Frag b;
#pragma unroll
      for (int s = 0; s < 16; ++s) b.h[s] = wb[(long)((s & 7) + 16 * (s >> 3)) * 128];
      acc0[t] = wmma16(a0.v, b.v, acc0[t]);
      acc1[t] = wmma16(a1.v, b.v, acc1[t]);
    }
  }

  // ---- scatter-add message rows into h by target node
  int cidx0[8], cidx1[8];
#pragma unroll
  for (int r = 0; r < 8; ++r) {
    long e2 = eBase + r + 8 * hh;
    long e3 = eBase + 16 + r + 8 * hh;
    cidx0[r] = (e2 < E) ? col[e2] : -1;
    cidx1[r] = (e3 < E) ? col[e3] : -1;
  }
#pragma unroll
  for (int t = 0; t < 8; ++t)
#pragma unroll
    for (int r = 0; r < 8; ++r) {
      if (cidx0[r] >= 0)
        atomAdd(hacc + (long)cidx0[r] * 128 + t * 16 + m, acc0[t][r]);
      if (cidx1[r] >= 0)
        atomAdd(hacc + (long)cidx1[r] * 128 + t * 16 + m, acc1[t][r]);
    }
}

// ------------------------------------------------- TAGConv fused matmul (WMMA)
// out = bias + sum_{k=0..3} h_k @ W_k  (feats: 4 f16 matrices [N][128],
// Wh: f16 [4][128][ncols]); optional ReLU.
__global__ void __launch_bounds__(256)
tag_matmul_k(const _Float16* __restrict__ feats, const _Float16* __restrict__ Wh,
             const float* __restrict__ bias, float* __restrict__ out,
             int nRows, int ncols, int relu) {
  const int lane = threadIdx.x & 31;
  const int wave = threadIdx.x >> 5;
  const int tile = blockIdx.x * 8 + wave;
  if (tile * 16 >= nRows) return;
  const int m = lane & 15;
  const int hh = lane >> 4;
  const long rowA = (long)(tile * 16 + m) * 128;

  // hoist all 16 A fragments (4 hop matrices x 4 K-chunks)
  v16h afrag[16];
#pragma unroll
  for (int mat = 0; mat < 4; ++mat) {
    const _Float16* fb = feats + (long)mat * (long)nRows * 128 + rowA;
#pragma unroll
    for (int c = 0; c < 4; ++c) {
      Frag a;
      a.f4[0] = *reinterpret_cast<const v4f*>(fb + c * 32 + hh * 8);
      a.f4[1] = *reinterpret_cast<const v4f*>(fb + c * 32 + 16 + hh * 8);
      afrag[mat * 4 + c] = a.v;
    }
  }

  const int nt = ncols >> 4;
  for (int t = 0; t < nt; ++t) {
    v8f acc = splat8(bias[t * 16 + m]);
#pragma unroll
    for (int mat = 0; mat < 4; ++mat) {
#pragma unroll
      for (int c = 0; c < 4; ++c) {
        const _Float16* wb = Wh + ((long)mat * 128 + c * 32 + 8 * hh) * ncols + t * 16 + m;
        Frag b;
#pragma unroll
        for (int s = 0; s < 16; ++s) b.h[s] = wb[(long)((s & 7) + 16 * (s >> 3)) * ncols];
        acc = wmma16(afrag[mat * 4 + c], b.v, acc);
      }
    }
#pragma unroll
    for (int r = 0; r < 8; ++r) {
      float v = acc[r];
      if (relu) v = fmaxf(v, 0.f);
      out[(long)(tile * 16 + r + 8 * hh) * ncols + t * 16 + m] = v;
    }
  }
}

// ----------------------------------------------------------------- launcher
extern "C" void kernel_launch(void* const* d_in, const int* in_sizes, int n_in,
                              void* d_out, int out_size, void* d_ws, size_t ws_size,
                              hipStream_t stream) {
  const float* x   = (const float*)d_in[0];
  const int*   ei  = (const int*)d_in[1];
  const float* ea  = (const float*)d_in[2];
  const float* W1  = (const float*)d_in[3];
  const float* b1  = (const float*)d_in[4];
  const float* W2  = (const float*)d_in[5];
  const float* b2  = (const float*)d_in[6];
  const float* t0W = (const float*)d_in[7];
  const float* t0b = (const float*)d_in[8];
  const float* t1W = (const float*)d_in[9];
  const float* t1b = (const float*)d_in[10];
  const float* t2W = (const float*)d_in[11];
  const float* t2b = (const float*)d_in[12];

  const int N = in_sizes[0] / 128;
  const int E = in_sizes[1] / 2;
  const int* row = ei;       // sources j
  const int* col = ei + E;   // targets i

  char* ws = (char*)d_ws;
  size_t off = 0;
  auto take = [&](size_t bytes) -> char* {
    char* p = ws + off;
    off = (off + bytes + 255) & ~(size_t)255;
    return p;
  };
  _Float16* w1h   = (_Float16*)take(288 * 128 * 2);
  _Float16* w2h   = (_Float16*)take(128 * 128 * 2);
  _Float16* t0h   = (_Float16*)take(4 * 128 * 128 * 2);
  _Float16* t1h   = (_Float16*)take(4 * 128 * 128 * 2);
  _Float16* t2h   = (_Float16*)take(4 * 128 * 64 * 2);
  _Float16* xh    = (_Float16*)take((size_t)N * 128 * 2);
  float*    bufA  = (float*)take((size_t)N * 128 * 4);
  float*    bufB  = (float*)take((size_t)N * 128 * 4);
  _Float16* hk    = (_Float16*)take((size_t)4 * N * 128 * 2);
  float*    degdis= (float*)take((size_t)N * 4);
  float*    nrm   = (float*)take((size_t)E * 4);

  auto cvt = [&](const float* in, _Float16* out, long n) {
    int n4 = (int)(n / 4);
    cvt_f16_k<<<(n4 + 255) / 256, 256, 0, stream>>>(in, out, n4);
  };

  // weight / feature conversions to f16
  hipMemsetAsync(bufA, 0, (size_t)N * 128 * 4, stream);
  hipMemsetAsync(degdis, 0, (size_t)N * 4, stream);
  pack_w1_k<<<(288 * 128 + 255) / 256, 256, 0, stream>>>(W1, w1h);
  cvt(W2, w2h, 128 * 128);
  cvt(t0W, t0h, 4 * 128 * 128);
  cvt(t1W, t1h, 4 * 128 * 128);
  cvt(t2W, t2h, 4 * 128 * 64);
  cvt(x, xh, (long)N * 128);

  // edge MLP + segment-sum into bufA (32 edges per wave, 4 waves per block)
  int etiles = (E + 31) / 32;
  edge_mlp_k<<<(etiles + 3) / 4, 128, 0, stream>>>(xh, ea, row, col, w1h, b1,
                                                   w2h, b2, bufA, E);

  // gcn norm
  deg_k<<<(E + 255) / 256, 256, 0, stream>>>(col, degdis, E);
  dis_k<<<(N + 255) / 256, 256, 0, stream>>>(degdis, N);
  norm_k<<<(E + 255) / 256, 256, 0, stream>>>(row, col, degdis, nrm, E);

  auto spmm = [&](const float* vin, float* vout) {
    hipMemsetAsync(vout, 0, (size_t)N * 128 * 4, stream);
    long th = (long)E * 32;
    spmm_k<<<(unsigned)((th + 255) / 256), 256, 0, stream>>>(row, col, nrm, vin,
                                                             vout, E);
  };

  const long NH = (long)N * 128;
  auto layer = [&](const _Float16* Wh, const float* bias, int ncols, int relu,
                   float* outp) {
    cvt(bufA, hk + 0 * NH, NH);                    // h0
    spmm(bufA, bufB); cvt(bufB, hk + 1 * NH, NH);  // h1
    spmm(bufB, bufA); cvt(bufA, hk + 2 * NH, NH);  // h2
    spmm(bufA, bufB); cvt(bufB, hk + 3 * NH, NH);  // h3
    int rtiles = (N + 15) / 16;
    tag_matmul_k<<<(rtiles + 7) / 8, 256, 0, stream>>>(hk, Wh, bias, outp, N,
                                                       ncols, relu);
  };

  layer(t0h, t0b, 128, 1, bufA);
  layer(t1h, t1b, 128, 1, bufA);
  layer(t2h, t2b, 64, 0, (float*)d_out);

  (void)n_in; (void)out_size; (void)ws_size;
}
